// TorchSOM_42494406427559
// MI455X (gfx1250) — matrix operand order repaired and checked by reading, not compiled, not run on hardware
//
#include <hip/hip_runtime.h>
#include <hip/hip_bf16.h>
#include <math.h>

#define X_DIM 128
#define Y_DIM 128
#define F_DIM 512
#define B_DIM 2048
#define XY    (X_DIM * Y_DIM)
#define LR    0.5f

typedef __attribute__((ext_vector_type(16))) __bf16 v16bf;
typedef __attribute__((ext_vector_type(8)))  float  v8f;

union FragBF { v16bf v; uint4 q[2]; };

__device__ __forceinline__ unsigned int fkey(float f) {
    unsigned int b = __float_as_uint(f);
    return (b & 0x80000000u) ? ~b : (b | 0x80000000u);
}

__device__ __forceinline__ unsigned long long shflxor16_u64(unsigned long long v, int m) {
    unsigned int lo = (unsigned int)v;
    unsigned int hi = (unsigned int)(v >> 32);
    lo = __shfl_xor(lo, m, 16);
    hi = __shfl_xor(hi, m, 16);
    return ((unsigned long long)hi << 32) | (unsigned long long)lo;
}

__device__ __forceinline__ void ld8(const float* p, float* d) {
    float4 a = *(const float4*)p;
    float4 b = *(const float4*)(p + 4);
    d[0]=a.x; d[1]=a.y; d[2]=a.z; d[3]=a.w;
    d[4]=b.x; d[5]=b.y; d[6]=b.z; d[7]=b.w;
}

__global__ void som_prep_data(const float* __restrict__ data,
                              __bf16* __restrict__ dbf,
                              __bf16* __restrict__ dT,
                              unsigned long long* __restrict__ packed) {
    int i = blockIdx.x * 256 + threadIdx.x;
    float v = data[i];
    dbf[i] = (__bf16)v;
    int b = i >> 9, f = i & 511;
    dT[f * B_DIM + b] = (__bf16)v;
    if (i < B_DIM) packed[i] = ~0ull;
}

__global__ void som_prep_w(const float* __restrict__ w,
                           __bf16* __restrict__ wbf,
                           float* __restrict__ wnorm2) {
    int n = blockIdx.x;
    int t = threadIdx.x;
    const float* row = w + n * F_DIM;
    float a0 = row[t], a1 = row[t + 256];
    wbf[n * F_DIM + t]       = (__bf16)a0;
    wbf[n * F_DIM + t + 256] = (__bf16)a1;
    float p = a0 * a0 + a1 * a1;
    #pragma unroll
    for (int m = 16; m >= 1; m >>= 1) p += __shfl_xor(p, m, 32);
    __shared__ float red[8];
    if ((t & 31) == 0) red[t >> 5] = p;
    __syncthreads();
    if (t == 0) {
        float s = 0.f;
        #pragma unroll
        for (int i = 0; i < 8; ++i) s += red[i];
        wnorm2[n] = s;
    }
}

__global__ void __launch_bounds__(256) som_dist_argmin(
        const __bf16* __restrict__ dbf,
        const __bf16* __restrict__ wbf,
        const float*  __restrict__ wnorm2,
        unsigned long long* __restrict__ packed) {

    const int lane = threadIdx.x & 31;
    const int wv   = threadIdx.x >> 5;
    const int h    = lane >> 4;
    const int ln   = lane & 15;

    const int mbase = blockIdx.x * 32;
    const int nbase = blockIdx.y * 512 + wv * 64;

    const __bf16* a0p = dbf + (size_t)(mbase + ln) * F_DIM + h * 8;
    const __bf16* a1p = a0p + 16 * F_DIM;
    const __bf16* bp0 = wbf + (size_t)(nbase +  0 + ln) * F_DIM + h * 8;
    const __bf16* bp1 = wbf + (size_t)(nbase + 16 + ln) * F_DIM + h * 8;
    const __bf16* bp2 = wbf + (size_t)(nbase + 32 + ln) * F_DIM + h * 8;
    const __bf16* bp3 = wbf + (size_t)(nbase + 48 + ln) * F_DIM + h * 8;

    v8f zero = {};
    v8f C[2][4];
    #pragma unroll
    for (int mt = 0; mt < 2; ++mt)
        #pragma unroll
        for (int nt = 0; nt < 4; ++nt) C[mt][nt] = zero;

    FragBF A0[2], A1[2], Bf[2][4];

    A0[0].q[0] = *(const uint4*)(a0p);       A0[0].q[1] = *(const uint4*)(a0p + 16);
    A1[0].q[0] = *(const uint4*)(a1p);       A1[0].q[1] = *(const uint4*)(a1p + 16);
    Bf[0][0].q[0] = *(const uint4*)(bp0);    Bf[0][0].q[1] = *(const uint4*)(bp0 + 16);
    Bf[0][1].q[0] = *(const uint4*)(bp1);    Bf[0][1].q[1] = *(const uint4*)(bp1 + 16);
    Bf[0][2].q[0] = *(const uint4*)(bp2);    Bf[0][2].q[1] = *(const uint4*)(bp2 + 16);
    Bf[0][3].q[0] = *(const uint4*)(bp3);    Bf[0][3].q[1] = *(const uint4*)(bp3 + 16);

    #pragma unroll
    for (int kk = 0; kk < 16; ++kk) {
        const int cur = kk & 1;
        const int nxt = cur ^ 1;
        if (kk < 15) {
            const int k = (kk + 1) * 32;
            A0[nxt].q[0] = *(const uint4*)(a0p + k);
            A0[nxt].q[1] = *(const uint4*)(a0p + k + 16);
            A1[nxt].q[0] = *(const uint4*)(a1p + k);
            A1[nxt].q[1] = *(const uint4*)(a1p + k + 16);
            Bf[nxt][0].q[0] = *(const uint4*)(bp0 + k);
            Bf[nxt][0].q[1] = *(const uint4*)(bp0 + k + 16);
            Bf[nxt][1].q[0] = *(const uint4*)(bp1 + k);
            Bf[nxt][1].q[1] = *(const uint4*)(bp1 + k + 16);
            Bf[nxt][2].q[0] = *(const uint4*)(bp2 + k);
            Bf[nxt][2].q[1] = *(const uint4*)(bp2 + k + 16);
            Bf[nxt][3].q[0] = *(const uint4*)(bp3 + k);
            Bf[nxt][3].q[1] = *(const uint4*)(bp3 + k + 16);
        }
        #pragma unroll
        for (int nt = 0; nt < 4; ++nt) {
            C[0][nt] = __builtin_amdgcn_wmma_f32_16x16x32_bf16(
                false, A0[cur].v, false, Bf[cur][nt].v, (short)0, C[0][nt], false, false);
            C[1][nt] = __builtin_amdgcn_wmma_f32_16x16x32_bf16(
                false, A1[cur].v, false, Bf[cur][nt].v, (short)0, C[1][nt], false, false);
        }
    }

    float wn[4];
    #pragma unroll
    for (int nt = 0; nt < 4; ++nt) wn[nt] = wnorm2[nbase + nt * 16 + ln];

    __shared__ unsigned long long red[32][8];

    #pragma unroll
    for (int mt = 0; mt < 2; ++mt) {
        #pragma unroll
        for (int r = 0; r < 8; ++r) {
            unsigned long long best = ~0ull;
            #pragma unroll
            for (int nt = 0; nt < 4; ++nt) {
                float s = wn[nt] - 2.0f * C[mt][nt][r];
                unsigned long long pk =
                    ((unsigned long long)fkey(s) << 32) |
                    (unsigned long long)(unsigned int)(nbase + nt * 16 + ln);
                best = best < pk ? best : pk;
            }
            #pragma unroll
            for (int m = 1; m < 16; m <<= 1) {
                unsigned long long o = shflxor16_u64(best, m);
                best = best < o ? best : o;
            }
            if (ln == 0) red[mt * 16 + h * 8 + r][wv] = best;
        }
    }
    __syncthreads();
    int t = threadIdx.x;
    if (t < 32) {
        unsigned long long best = red[t][0];
        #pragma unroll
        for (int i = 1; i < 8; ++i) {
            unsigned long long o = red[t][i];
            best = best < o ? best : o;
        }
        atomicMin(&packed[mbase + t], best);
    }
}

__global__ void som_neigh_tables(const unsigned long long* __restrict__ packed,
                                 float* __restrict__ Rx,
                                 float* __restrict__ Cy) {
    int b = blockIdx.x;
    int x = threadIdx.x;
    unsigned int idx = (unsigned int)(packed[b] & 0xFFFFFFFFu);
    float row = (float)(idx >> 7);
    float col = (float)(idx & 127u);
    float dr = (float)x - row;
    float dc = (float)x - col;
    Rx[x * B_DIM + b] = __expf(-0.5f * dr * dr);
    Cy[x * B_DIM + b] = __expf(-0.5f * dc * dc);
}

__global__ void __launch_bounds__(256) som_update(
        const float* __restrict__ Rx,
        const float* __restrict__ Cy,
        const __bf16* __restrict__ dT,
        const float* __restrict__ w,
        float* __restrict__ out) {

    const int lane = threadIdx.x & 31;
    const int wv   = threadIdx.x >> 5;
    const int h    = lane >> 4;
    const int ln   = lane & 15;
    const int mgrp = wv >> 2;
    const int nblk = wv & 3;

    const int wmbase = blockIdx.x * 64 + mgrp * 32;
    const int nbase  = blockIdx.y * 256 + nblk * 64;

    v8f zero = {};
    v8f C[2][4];
    #pragma unroll
    for (int mt = 0; mt < 2; ++mt)
        #pragma unroll
        for (int nt = 0; nt < 4; ++nt) C[mt][nt] = zero;

    const int m0 = wmbase + ln;
    const int m1 = m0 + 16;
    const float* rx0 = Rx + (size_t)(m0 >> 7)  * B_DIM + h * 8;
    const float* cy0 = Cy + (size_t)(m0 & 127) * B_DIM + h * 8;
    const float* rx1 = Rx + (size_t)(m1 >> 7)  * B_DIM + h * 8;
    const float* cy1 = Cy + (size_t)(m1 & 127) * B_DIM + h * 8;

    const __bf16* bp0 = dT + (size_t)(nbase +  0 + ln) * B_DIM + h * 8;
    const __bf16* bp1 = dT + (size_t)(nbase + 16 + ln) * B_DIM + h * 8;
    const __bf16* bp2 = dT + (size_t)(nbase + 32 + ln) * B_DIM + h * 8;
    const __bf16* bp3 = dT + (size_t)(nbase + 48 + ln) * B_DIM + h * 8;

    float nsum0 = 0.f, nsum1 = 0.f;
    FragBF A0, A1, Bf[2][4];

    {
        float rr[16], cc[16];
        ld8(rx0, rr); ld8(rx0 + 16, rr + 8);
        ld8(cy0, cc); ld8(cy0 + 16, cc + 8);
        #pragma unroll
        for (int j = 0; j < 16; ++j) { float p = rr[j]*cc[j]; nsum0 += p; A0.v[j] = (__bf16)p; }
        ld8(rx1, rr); ld8(rx1 + 16, rr + 8);
        ld8(cy1, cc); ld8(cy1 + 16, cc + 8);
        #pragma unroll
        for (int j = 0; j < 16; ++j) { float p = rr[j]*cc[j]; nsum1 += p; A1.v[j] = (__bf16)p; }
        Bf[0][0].q[0] = *(const uint4*)(bp0);  Bf[0][0].q[1] = *(const uint4*)(bp0 + 16);
        Bf[0][1].q[0] = *(const uint4*)(bp1);  Bf[0][1].q[1] = *(const uint4*)(bp1 + 16);
        Bf[0][2].q[0] = *(const uint4*)(bp2);  Bf[0][2].q[1] = *(const uint4*)(bp2 + 16);
        Bf[0][3].q[0] = *(const uint4*)(bp3);  Bf[0][3].q[1] = *(const uint4*)(bp3 + 16);
    }

    #pragma unroll 2
    for (int kk = 0; kk < 64; ++kk) {
        const int cur = kk & 1;
        const int nxt = cur ^ 1;
        const bool pf = (kk < 63);
        const int k  = (kk + 1) * 32;

        float rr0[16], cc0[16], rr1[16], cc1[16];
        if (pf) {
            ld8(rx0 + k, rr0); ld8(rx0 + k + 16, rr0 + 8);
            ld8(cy0 + k, cc0); ld8(cy0 + k + 16, cc0 + 8);
            ld8(rx1 + k, rr1); ld8(rx1 + k + 16, rr1 + 8);
            ld8(cy1 + k, cc1); ld8(cy1 + k + 16, cc1 + 8);
            Bf[nxt][0].q[0] = *(const uint4*)(bp0 + k);
            Bf[nxt][0].q[1] = *(const uint4*)(bp0 + k + 16);
            Bf[nxt][1].q[0] = *(const uint4*)(bp1 + k);
            Bf[nxt][1].q[1] = *(const uint4*)(bp1 + k + 16);
            Bf[nxt][2].q[0] = *(const uint4*)(bp2 + k);
            Bf[nxt][2].q[1] = *(const uint4*)(bp2 + k + 16);
            Bf[nxt][3].q[0] = *(const uint4*)(bp3 + k);
            Bf[nxt][3].q[1] = *(const uint4*)(bp3 + k + 16);
        }

        #pragma unroll
        for (int nt = 0; nt < 4; ++nt) {
            C[0][nt] = __builtin_amdgcn_wmma_f32_16x16x32_bf16(
                false, A0.v, false, Bf[cur][nt].v, (short)0, C[0][nt], false, false);
            C[1][nt] = __builtin_amdgcn_wmma_f32_16x16x32_bf16(
                false, A1.v, false, Bf[cur][nt].v, (short)0, C[1][nt], false, false);
        }

        if (pf) {
            #pragma unroll
            for (int j = 0; j < 16; ++j) {
                float p0 = rr0[j] * cc0[j]; nsum0 += p0; A0.v[j] = (__bf16)p0;
                float p1 = rr1[j] * cc1[j]; nsum1 += p1; A1.v[j] = (__bf16)p1;
            }
        }
    }

    nsum0 += __shfl_xor(nsum0, 16, 32);
    nsum1 += __shfl_xor(nsum1, 16, 32);
    __shared__ float nmean[64];
    if (nblk == 0 && lane < 16) {
        nmean[mgrp * 32 + lane]      = nsum0 * (1.0f / B_DIM);
        nmean[mgrp * 32 + 16 + lane] = nsum1 * (1.0f / B_DIM);
    }
    __syncthreads();

    const float invB = 1.0f / B_DIM;
    #pragma unroll
    for (int mt = 0; mt < 2; ++mt) {
        #pragma unroll
        for (int t = 0; t < 4; ++t) {
            #pragma unroll
            for (int r = 0; r < 8; ++r) {
                int ml  = mgrp * 32 + mt * 16 + h * 8 + r;
                int mm  = blockIdx.x * 64 + ml;
                int col = nbase + t * 16 + ln;
                size_t addr = (size_t)mm * F_DIM + col;
                float wvv = w[addr];
                float nd  = C[mt][t][r] * invB;
                out[addr] = wvv + LR * (nd - nmean[ml] * wvv);
            }
        }
    }
}

extern "C" void kernel_launch(void* const* d_in, const int* in_sizes, int n_in,
                              void* d_out, int out_size, void* d_ws, size_t ws_size,
                              hipStream_t stream) {
    const float* data    = (const float*)d_in[0];
    const float* weights = (const float*)d_in[1];
    float* out = (float*)d_out;
    char* ws = (char*)d_ws;

    unsigned long long* packed = (unsigned long long*)(ws);
    float*  wnorm2 = (float*)(ws + (64u  << 10));
    float*  Rx     = (float*)(ws + (128u << 10));
    float*  Cy     = (float*)(ws + (128u << 10) + (1u << 20));
    __bf16* dbf    = (__bf16*)(ws + (128u << 10) + (2u << 20));
    __bf16* dT     = (__bf16*)(ws + (128u << 10) + (4u << 20));
    __bf16* wbf    = (__bf16*)(ws + (128u << 10) + (6u << 20));

    som_prep_data<<<(B_DIM * F_DIM) / 256, 256, 0, stream>>>(data, dbf, dT, packed);
    som_prep_w<<<XY, 256, 0, stream>>>(weights, wbf, wnorm2);
    som_dist_argmin<<<dim3(B_DIM / 32, XY / 512), 256, 0, stream>>>(dbf, wbf, wnorm2, packed);
    som_neigh_tables<<<B_DIM, X_DIM, 0, stream>>>(packed, Rx, Cy);
    som_update<<<dim3(XY / 64, F_DIM / 256), 256, 0, stream>>>(Rx, Cy, dT, weights, out);
}
